// MixtralSparseMoeBlock_15779709845991
// MI455X (gfx1250) — matrix operand order repaired and checked by reading, not compiled, not run on hardware
//
#include <hip/hip_runtime.h>
#include <hip/hip_bf16.h>
#include <stdint.h>

typedef __attribute__((ext_vector_type(16))) __bf16 v16bf;
typedef __attribute__((ext_vector_type(8)))  float  v8f;
typedef __attribute__((ext_vector_type(4)))  unsigned int v4u;
typedef __attribute__((ext_vector_type(8)))  int v8i;
typedef __attribute__((ext_vector_type(4)))  int v4i;

#define NT 2048   // tokens (B*S)
#define NH 1024   // hidden
#define NI 2048   // intermediate
#define NE 8      // experts
#define TM 128    // token tile
#define TN 128    // output-col tile
#define TK 32     // K step (matches wmma bf16 K)
#define AKP 40    // padded LDS row stride (elements): 64B row + 16B pad -> bank-conflict-free
#define BKP 40

// ---------------------------------------------------------------------------
// TDM async 2-D bf16 tile load, global row-major -> LDS with 16B pad per 64B
// row (pad_enable, interval=16 DW, amount=4 DW) so the tile lands directly in
// the AKP=40-element padded layout.  D# per CDNA5 ISA 8.3/8.4.
// ---------------------------------------------------------------------------
__device__ __forceinline__ void tdm_load_tile_bf16(uint32_t lds_off, const __bf16* g,
                                                   uint32_t row_len, uint32_t nrows)
{
  uint64_t ga = (uint64_t)(uintptr_t)g;
  v4u g0;
  g0[0] = 1u;                                                   // count=1, user D#
  g0[1] = lds_off;                                              // LDS dest (bytes)
  g0[2] = (uint32_t)(ga & 0xffffffffu);                         // global_addr[31:0]
  g0[3] = ((uint32_t)(ga >> 32) & 0x01ffffffu) | 0x80000000u;   // addr[56:32]|type=2
  v8i g1;
  // data_size=2B (code 1) | pad_enable | pad_interval=16DW (code 3) | pad_amount=4DW (code 3)
  g1[0] = (int)((1u << 16) | (1u << 20) | (3u << 22) | (3u << 25));
  g1[1] = (int)((row_len & 0xffffu) << 16);                     // tensor_dim0 lo16
  g1[2] = (int)(((row_len >> 16) & 0xffffu) | ((nrows & 0xffffu) << 16));
  g1[3] = (int)(((nrows >> 16) & 0xffffu) | ((uint32_t)TK << 16)); // tile_dim0=32
  g1[4] = (int)TM;                                              // tile_dim1=128, dim2=0
  g1[5] = (int)row_len;                                         // dim0_stride[31:0]
  g1[6] = 0;
  g1[7] = 0;
  v4i z4; z4[0] = 0; z4[1] = 0; z4[2] = 0; z4[3] = 0;
  v8i z8;
#pragma unroll
  for (int i = 0; i < 8; ++i) z8[i] = 0;
  __builtin_amdgcn_tensor_load_to_lds(g0, g1, z4, z4, z8, 0);
}

// Two 16B LDS reads -> one 16-element bf16 fragment (ds_load_b128 x2)
__device__ __forceinline__ v16bf ld_frag2(const __bf16* p0, const __bf16* p1)
{
  union { uint4 q[2]; v16bf v; } u;
  u.q[0] = *(const uint4*)p0;
  u.q[1] = *(const uint4*)p1;
  return u.v;
}

// ---------------------------------------------------------------------------
// x (fp32) -> xb (bf16)
// ---------------------------------------------------------------------------
__global__ __launch_bounds__(256) void cvt_x_kernel(const float* __restrict__ x,
                                                    __bf16* __restrict__ xb)
{
  int i = (blockIdx.x * 256 + threadIdx.x) * 4;
  float4 v = *(const float4*)(x + i);
  xb[i + 0] = (__bf16)v.x;
  xb[i + 1] = (__bf16)v.y;
  xb[i + 2] = (__bf16)v.z;
  xb[i + 3] = (__bf16)v.w;
}

// ---------------------------------------------------------------------------
// Router: one wave per token; top-2 softmax -> dense rw[t, 0..7]
// ---------------------------------------------------------------------------
__global__ __launch_bounds__(256) void router_kernel(const float* __restrict__ x,
                                                     const float* __restrict__ Wg,
                                                     float* __restrict__ rw)
{
  const int lane = threadIdx.x & 31;
  const int wv   = threadIdx.x >> 5;
  const int t    = blockIdx.x * 8 + wv;

  float acc[NE];
#pragma unroll
  for (int e = 0; e < NE; ++e) acc[e] = 0.f;

  for (int h = lane; h < NH; h += 32) {
    float xv = x[(size_t)t * NH + h];
#pragma unroll
    for (int e = 0; e < NE; ++e) acc[e] += xv * Wg[h * NE + e];
  }
#pragma unroll
  for (int off = 16; off > 0; off >>= 1) {
#pragma unroll
    for (int e = 0; e < NE; ++e) acc[e] += __shfl_xor(acc[e], off, 32);
  }
  if (lane == 0) {
    int i0 = 0; float v0 = acc[0];
#pragma unroll
    for (int e = 1; e < NE; ++e) { if (acc[e] > v0) { v0 = acc[e]; i0 = e; } }
    int i1 = (i0 == 0) ? 1 : 0; float v1 = acc[i1];
#pragma unroll
    for (int e = 0; e < NE; ++e) { if (e != i0 && acc[e] > v1) { v1 = acc[e]; i1 = e; } }
    float p1 = __expf(v1 - v0);
    float z  = 1.f + p1;
#pragma unroll
    for (int e = 0; e < NE; ++e) rw[(size_t)t * NE + e] = 0.f;
    rw[(size_t)t * NE + i0] = 1.f / z;
    rw[(size_t)t * NE + i1] = p1 / z;
  }
}

// ---------------------------------------------------------------------------
// Up/gate GEMM + SiLU + routing weight (double-buffered, TDM for A,
// transposed bf16 B tiles in LDS for contiguous b128 fragment loads).
// ---------------------------------------------------------------------------
__global__ __launch_bounds__(256)
void moe_up_kernel(const __bf16* __restrict__ xb,
                   const float* __restrict__ W1,
                   const float* __restrict__ W3,
                   const float* __restrict__ rw,
                   __bf16* __restrict__ act, int e)
{
  __shared__ __bf16 As[2][TM * AKP];
  __shared__ __bf16 B1s[2][TN * BKP];
  __shared__ __bf16 B3s[2][TN * BKP];

  const int tid  = threadIdx.x;
  const int lane = tid & 31;
  const int wv   = tid >> 5;
  const int wm   = wv & 3;
  const int wn   = wv >> 2;
  const int bm   = blockIdx.y, bn = blockIdx.x;

  const float* W1e = W1 + (size_t)e * NH * NI + (size_t)bn * TN;
  const float* W3e = W3 + (size_t)e * NH * NI + (size_t)bn * TN;
  const __bf16* arow = xb + (size_t)bm * TM * NH;

  const int hsel = lane >> 4;
  const int l15  = lane & 15;
  const int srow = tid >> 6;            // staging: 4 rows per pass (tid/64)
  const int scol = (tid & 63) << 1;     // 2 elements per thread per row? -- see below

  v8f cu[2][4], cg[2][4];
#pragma unroll
  for (int i = 0; i < 2; ++i)
#pragma unroll
    for (int j = 0; j < 4; ++j)
#pragma unroll
      for (int r = 0; r < 8; ++r) { cu[i][j][r] = 0.f; cg[i][j][r] = 0.f; }

  // staging coords: thread handles float4 slots l = j*256+tid, row=l>>5, col=(l&31)*4
  const int nk = NH / TK;

  // ---- prologue: tile 0 -> buffer 0
  if (wv == 0)
    tdm_load_tile_bf16((uint32_t)(uintptr_t)&As[0][0], arow, NH, NT);
  {
    float4 r1[4], r3[4];
#pragma unroll
    for (int j = 0; j < 4; ++j) {
      int l = j * 256 + tid, row = l >> 5, col = (l & 31) << 2;
      r1[j] = *(const float4*)(W1e + (size_t)row * NI + col);
      r3[j] = *(const float4*)(W3e + (size_t)row * NI + col);
    }
#pragma unroll
    for (int j = 0; j < 4; ++j) {
      int l = j * 256 + tid, row = l >> 5, col = (l & 31) << 2;
      B1s[0][(col + 0) * BKP + row] = (__bf16)r1[j].x;
      B1s[0][(col + 1) * BKP + row] = (__bf16)r1[j].y;
      B1s[0][(col + 2) * BKP + row] = (__bf16)r1[j].z;
      B1s[0][(col + 3) * BKP + row] = (__bf16)r1[j].w;
      B3s[0][(col + 0) * BKP + row] = (__bf16)r3[j].x;
      B3s[0][(col + 1) * BKP + row] = (__bf16)r3[j].y;
      B3s[0][(col + 2) * BKP + row] = (__bf16)r3[j].z;
      B3s[0][(col + 3) * BKP + row] = (__bf16)r3[j].w;
    }
  }
  if (wv == 0) __builtin_amdgcn_s_wait_tensorcnt(0);
  __syncthreads();

  for (int kk = 0; kk < nk; ++kk) {
    const int cur = kk & 1, nxt = cur ^ 1;
    const bool has_next = (kk + 1) < nk;

    float4 r1[4], r3[4];
    if (has_next) {
      if (wv == 0)
        tdm_load_tile_bf16((uint32_t)(uintptr_t)&As[nxt][0],
                           arow + (size_t)(kk + 1) * TK, NH, NT);
#pragma unroll
      for (int j = 0; j < 4; ++j) {
        int l = j * 256 + tid, row = l >> 5, col = (l & 31) << 2;
        const float* p1 = W1e + (size_t)((kk + 1) * TK + row) * NI + col;
        const float* p3 = W3e + (size_t)((kk + 1) * TK + row) * NI + col;
        r1[j] = *(const float4*)p1;
        r3[j] = *(const float4*)p3;
        if (kk + 2 < nk) {
          __builtin_prefetch(p1 + (size_t)TK * NI, 0, 1);   // global_prefetch_b8
          __builtin_prefetch(p3 + (size_t)TK * NI, 0, 1);
        }
      }
    }

    // ---- compute current tile: contiguous b128 fragment loads
    v16bf af[2];
#pragma unroll
    for (int mt = 0; mt < 2; ++mt) {
      const int m = wm * 32 + mt * 16 + l15;
      af[mt] = ld_frag2(&As[cur][m * AKP + hsel * 8],
                        &As[cur][m * AKP + 16 + hsel * 8]);
    }
#pragma unroll
    for (int nt = 0; nt < 4; ++nt) {
      const int n = wn * 64 + nt * 16 + l15;
      v16bf b1f = ld_frag2(&B1s[cur][n * BKP + hsel * 8],
                           &B1s[cur][n * BKP + 16 + hsel * 8]);
      v16bf b3f = ld_frag2(&B3s[cur][n * BKP + hsel * 8],
                           &B3s[cur][n * BKP + 16 + hsel * 8]);
#pragma unroll
      for (int mt = 0; mt < 2; ++mt) {
        cu[mt][nt] = __builtin_amdgcn_wmma_f32_16x16x32_bf16(
            false, af[mt], false, b1f, (short)0, cu[mt][nt], false, false);
        cg[mt][nt] = __builtin_amdgcn_wmma_f32_16x16x32_bf16(
            false, af[mt], false, b3f, (short)0, cg[mt][nt], false, false);
      }
    }

    if (has_next) {
#pragma unroll
      for (int j = 0; j < 4; ++j) {
        int l = j * 256 + tid, row = l >> 5, col = (l & 31) << 2;
        B1s[nxt][(col + 0) * BKP + row] = (__bf16)r1[j].x;
        B1s[nxt][(col + 1) * BKP + row] = (__bf16)r1[j].y;
        B1s[nxt][(col + 2) * BKP + row] = (__bf16)r1[j].z;
        B1s[nxt][(col + 3) * BKP + row] = (__bf16)r1[j].w;
        B3s[nxt][(col + 0) * BKP + row] = (__bf16)r3[j].x;
        B3s[nxt][(col + 1) * BKP + row] = (__bf16)r3[j].y;
        B3s[nxt][(col + 2) * BKP + row] = (__bf16)r3[j].z;
        B3s[nxt][(col + 3) * BKP + row] = (__bf16)r3[j].w;
      }
      if (wv == 0) __builtin_amdgcn_s_wait_tensorcnt(0);
      __syncthreads();
    }
  }

  // ---- epilogue: act = rw * silu(up) * gate  (bf16)
#pragma unroll
  for (int mt = 0; mt < 2; ++mt)
#pragma unroll
    for (int nt = 0; nt < 4; ++nt) {
      const int i = bn * TN + wn * 64 + nt * 16 + l15;
#pragma unroll
      for (int r = 0; r < 8; ++r) {
        const int t = bm * TM + wm * 32 + mt * 16 + hsel * 8 + r;
        float up = cu[mt][nt][r];
        float gt = cg[mt][nt][r];
        float s  = up / (1.f + __expf(-up));
        float a  = s * gt * rw[(size_t)t * NE + e];
        act[(size_t)t * NI + i] = (__bf16)a;
      }
    }
}

// ---------------------------------------------------------------------------
// Down GEMM: out (+)= act_e @ W2_e   (same pipeline, single B matrix)
// ---------------------------------------------------------------------------
__global__ __launch_bounds__(256)
void moe_down_kernel(const __bf16* __restrict__ act,
                     const float* __restrict__ W2,
                     float* __restrict__ out, int e, int first)
{
  __shared__ __bf16 As[2][TM * AKP];
  __shared__ __bf16 Bs[2][TN * BKP];

  const int tid  = threadIdx.x;
  const int lane = tid & 31;
  const int wv   = tid >> 5;
  const int wm   = wv & 3;
  const int wn   = wv >> 2;
  const int bm   = blockIdx.y, bn = blockIdx.x;

  const float* W2e = W2 + (size_t)e * NI * NH + (size_t)bn * TN;
  const __bf16* arow = act + (size_t)bm * TM * NI;

  const int hsel = lane >> 4;
  const int l15  = lane & 15;

  v8f c[2][4];
#pragma unroll
  for (int i = 0; i < 2; ++i)
#pragma unroll
    for (int j = 0; j < 4; ++j)
#pragma unroll
      for (int r = 0; r < 8; ++r) c[i][j][r] = 0.f;

  const int nk = NI / TK;

  if (wv == 0)
    tdm_load_tile_bf16((uint32_t)(uintptr_t)&As[0][0], arow, NI, NT);
  {
    float4 rb[4];
#pragma unroll
    for (int j = 0; j < 4; ++j) {
      int l = j * 256 + tid, row = l >> 5, col = (l & 31) << 2;
      rb[j] = *(const float4*)(W2e + (size_t)row * NH + col);
    }
#pragma unroll
    for (int j = 0; j < 4; ++j) {
      int l = j * 256 + tid, row = l >> 5, col = (l & 31) << 2;
      Bs[0][(col + 0) * BKP + row] = (__bf16)rb[j].x;
      Bs[0][(col + 1) * BKP + row] = (__bf16)rb[j].y;
      Bs[0][(col + 2) * BKP + row] = (__bf16)rb[j].z;
      Bs[0][(col + 3) * BKP + row] = (__bf16)rb[j].w;
    }
  }
  if (wv == 0) __builtin_amdgcn_s_wait_tensorcnt(0);
  __syncthreads();

  for (int kk = 0; kk < nk; ++kk) {
    const int cur = kk & 1, nxt = cur ^ 1;
    const bool has_next = (kk + 1) < nk;

    float4 rb[4];
    if (has_next) {
      if (wv == 0)
        tdm_load_tile_bf16((uint32_t)(uintptr_t)&As[nxt][0],
                           arow + (size_t)(kk + 1) * TK, NI, NT);
#pragma unroll
      for (int j = 0; j < 4; ++j) {
        int l = j * 256 + tid, row = l >> 5, col = (l & 31) << 2;
        const float* p2 = W2e + (size_t)((kk + 1) * TK + row) * NH + col;
        rb[j] = *(const float4*)p2;
        if (kk + 2 < nk) __builtin_prefetch(p2 + (size_t)TK * NH, 0, 1);
      }
    }

    v16bf af[2];
#pragma unroll
    for (int mt = 0; mt < 2; ++mt) {
      const int m = wm * 32 + mt * 16 + l15;
      af[mt] = ld_frag2(&As[cur][m * AKP + hsel * 8],
                        &As[cur][m * AKP + 16 + hsel * 8]);
    }
#pragma unroll
    for (int nt = 0; nt < 4; ++nt) {
      const int n = wn * 64 + nt * 16 + l15;
      v16bf bf = ld_frag2(&Bs[cur][n * BKP + hsel * 8],
                          &Bs[cur][n * BKP + 16 + hsel * 8]);
#pragma unroll
      for (int mt = 0; mt < 2; ++mt)
        c[mt][nt] = __builtin_amdgcn_wmma_f32_16x16x32_bf16(
            false, af[mt], false, bf, (short)0, c[mt][nt], false, false);
    }

    if (has_next) {
#pragma unroll
      for (int j = 0; j < 4; ++j) {
        int l = j * 256 + tid, row = l >> 5, col = (l & 31) << 2;
        Bs[nxt][(col + 0) * BKP + row] = (__bf16)rb[j].x;
        Bs[nxt][(col + 1) * BKP + row] = (__bf16)rb[j].y;
        Bs[nxt][(col + 2) * BKP + row] = (__bf16)rb[j].z;
        Bs[nxt][(col + 3) * BKP + row] = (__bf16)rb[j].w;
      }
      if (wv == 0) __builtin_amdgcn_s_wait_tensorcnt(0);
      __syncthreads();
    }
  }

#pragma unroll
  for (int mt = 0; mt < 2; ++mt)
#pragma unroll
    for (int nt = 0; nt < 4; ++nt) {
      const int h = bn * TN + wn * 64 + nt * 16 + l15;
#pragma unroll
      for (int r = 0; r < 8; ++r) {
        const int t = bm * TM + wm * 32 + mt * 16 + hsel * 8 + r;
        size_t idx = (size_t)t * NH + h;
        float v = c[mt][nt][r];
        if (!first) v += out[idx];
        out[idx] = v;
      }
    }
}

// ---------------------------------------------------------------------------
extern "C" void kernel_launch(void* const* d_in, const int* in_sizes, int n_in,
                              void* d_out, int out_size, void* d_ws, size_t ws_size,
                              hipStream_t stream)
{
  (void)in_sizes; (void)n_in; (void)out_size; (void)ws_size;
  const float* x  = (const float*)d_in[0];   // (1, 2048, 1024)
  const float* Wg = (const float*)d_in[1];   // (1024, 8)
  const float* W1 = (const float*)d_in[2];   // (8, 1024, 2048)
  const float* W2 = (const float*)d_in[3];   // (8, 2048, 1024)
  const float* W3 = (const float*)d_in[4];   // (8, 1024, 2048)
  float* out = (float*)d_out;                // (1, 2048, 1024)

  // Workspace: xb 4 MiB | rw 64 KiB | act 8 MiB
  char*   ws  = (char*)d_ws;
  __bf16* xb  = (__bf16*)ws;
  float*  rw  = (float*)(ws + (size_t)NT * NH * 2);
  __bf16* act = (__bf16*)(ws + (size_t)NT * NH * 2 + (size_t)NT * NE * 4);

  cvt_x_kernel<<<dim3((NT * NH) / 1024), 256, 0, stream>>>(x, xb);
  router_kernel<<<dim3(NT / 8), 256, 0, stream>>>(x, Wg, rw);

  for (int e = 0; e < NE; ++e) {
    moe_up_kernel<<<dim3(NI / TN, NT / TM), 256, 0, stream>>>(xb, W1, W3, rw, act, e);
    moe_down_kernel<<<dim3(NH / TN, NT / TM), 256, 0, stream>>>(act, W2, out, e, e == 0);
  }
}
